// LightGroupAttnBlock_11433202942252
// MI455X (gfx1250) — matrix operand order repaired and checked by reading, not compiled, not run on hardware
//
#include <hip/hip_runtime.h>
#include <hip/hip_bf16.h>

// ---------------------------------------------------------------------------
// Problem constants (from the reference)
// ---------------------------------------------------------------------------
constexpr int B_  = 8;
constexpr int NQ_ = 256;
constexpr int NK_ = 4096;
constexpr int C_  = 1024;
constexpr int H_  = 16;
constexpr int D_  = 64;          // head dim
constexpr float EPS_   = 1e-5f;
constexpr float SCALE_ = 0.125f; // 1/sqrt(64)

// ---------------------------------------------------------------------------
// WMMA fragment helpers (CDNA5 V_WMMA_F32_16X16X32_F16, wave32)
// ---------------------------------------------------------------------------
typedef __attribute__((ext_vector_type(16))) _Float16 v16h;
typedef __attribute__((ext_vector_type(8)))  _Float16 v8h;
typedef __attribute__((ext_vector_type(8)))  float    v8f;

// int4 vector with explicit address spaces for the async-copy builtin:
// hipcc reports param 1 as "int __vector(4) __device__ *"  (AS1 = global).
typedef __attribute__((ext_vector_type(4))) int i4_t;
typedef __attribute__((address_space(1))) i4_t gi4_t;   // global int4
typedef __attribute__((address_space(3))) i4_t li4_t;   // LDS int4
typedef __attribute__((address_space(3))) void LAS;     // LDS void (asm path)

union HFrag { v16h v; v8h h[2]; };

// A-matrix 16x32 f16: lane holds row M=lane&15; halfs [0..7] at K=koff,
// halfs [8..15] at K=koff+16, koff=(lane>>4)*8. Caller pre-offsets by koff.
__device__ __forceinline__ v16h load_fragA(const _Float16* p) {
  HFrag f;
  f.h[0] = *(const v8h*)(p);
  f.h[1] = *(const v8h*)(p + 16);
  return f.v;
}

// B-matrix 32x16 f16: lane holds column N=lane&15; halfs [0..7] at K=koff,
// halfs [8..15] at K=koff+16, koff=(lane>>4)*8. Caller pre-offsets by koff.
__device__ __forceinline__ v16h load_fragB(const _Float16* p) {
  HFrag f;
  f.h[0] = *(const v8h*)(p);
  f.h[1] = *(const v8h*)(p + 16);
  return f.v;
}

__device__ __forceinline__ v8f wmma_f16(v16h a, v16h b, v8f c) {
  return __builtin_amdgcn_wmma_f32_16x16x32_f16(
      /*neg_a=*/false, a, /*neg_b=*/false, b,
      /*c_mod=*/(short)0, c, /*reuse_a=*/false, /*reuse_b=*/false);
}

// ---------------------------------------------------------------------------
// CDNA5 async global->LDS copy (ASYNCcnt path). 16 bytes per lane.
// ---------------------------------------------------------------------------
__device__ __forceinline__ void async_copy16(const _Float16* g, _Float16* l) {
#if __has_builtin(__builtin_amdgcn_global_load_async_to_lds_b128)
  __builtin_amdgcn_global_load_async_to_lds_b128(
      (gi4_t*)(_Float16*)g, (li4_t*)l, 0, 0);
#else
  unsigned loff = (unsigned)(unsigned long long)(LAS*)l;
  asm volatile("global_load_async_to_lds_b128 %0, %1, off"
               :: "v"(loff), "v"((unsigned long long)g) : "memory");
#endif
}

#if __has_builtin(__builtin_amdgcn_s_wait_asynccnt)
#define WAIT_ASYNC(n) __builtin_amdgcn_s_wait_asynccnt(n)
#else
#define WAIT_ASYNC(n) asm volatile("s_wait_asynccnt %0" :: "n"(n) : "memory")
#endif

// ---------------------------------------------------------------------------
// Reductions (wave32)
// ---------------------------------------------------------------------------
__device__ __forceinline__ float wave_sum(float v) {
  #pragma unroll
  for (int o = 16; o > 0; o >>= 1) v += __shfl_xor(v, o, 32);
  return v;
}
__device__ __forceinline__ float wave_max(float v) {
  #pragma unroll
  for (int o = 16; o > 0; o >>= 1) v = fmaxf(v, __shfl_xor(v, o, 32));
  return v;
}

// 256-thread block reduction (8 waves)
__device__ __forceinline__ float block_sum(float v, float* scratch) {
  const int lane = threadIdx.x & 31;
  const int wid  = threadIdx.x >> 5;
  v = wave_sum(v);
  if (lane == 0) scratch[wid] = v;
  __syncthreads();
  float r = (threadIdx.x < 8) ? scratch[threadIdx.x] : 0.f;
  if (wid == 0) {
    r = wave_sum(r);
    if (lane == 0) scratch[0] = r;
  }
  __syncthreads();
  float res = scratch[0];
  __syncthreads();
  return res;
}

// ---------------------------------------------------------------------------
// Kernel 1: LayerNorm(query) + query_pos -> f16   (one block per row of 1024)
// ---------------------------------------------------------------------------
__global__ void prep_q_kernel(const float* __restrict__ x,
                              const float* __restrict__ pos,
                              const float* __restrict__ w,
                              const float* __restrict__ bb,
                              _Float16* __restrict__ out) {
  const int row = blockIdx.x;
  const size_t base = (size_t)row * C_;
  float4 v = ((const float4*)(x + base))[threadIdx.x];
  float arr[4] = {v.x, v.y, v.z, v.w};

  __shared__ float s1[8], s2[8];
  float s  = arr[0] + arr[1] + arr[2] + arr[3];
  float ss = arr[0]*arr[0] + arr[1]*arr[1] + arr[2]*arr[2] + arr[3]*arr[3];
  float mu  = block_sum(s,  s1) * (1.0f / C_);
  float var = block_sum(ss, s2) * (1.0f / C_) - mu * mu;
  float rs  = rsqrtf(var + EPS_);

  const int i0 = threadIdx.x * 4;
  #pragma unroll
  for (int j = 0; j < 4; ++j) {
    int c = i0 + j;
    float y = (arr[j] - mu) * rs * w[c] + bb[c] + pos[base + c];
    out[base + c] = (_Float16)y;
  }
}

// ---------------------------------------------------------------------------
// Kernel 2: LayerNorm(memory): xk = kn+mem_pos (f16, row major) and
//           Vt = kn transposed per head: [b][h][d][k] (f16)
// ---------------------------------------------------------------------------
__global__ void prep_k_kernel(const float* __restrict__ x,
                              const float* __restrict__ pos,
                              const float* __restrict__ w,
                              const float* __restrict__ bb,
                              _Float16* __restrict__ xk,
                              _Float16* __restrict__ vt) {
  const int row  = blockIdx.x;       // 0 .. B*NK-1
  const int bidx = row / NK_;
  const int kk   = row - bidx * NK_;
  const size_t base = (size_t)row * C_;
  float4 v = ((const float4*)(x + base))[threadIdx.x];
  float arr[4] = {v.x, v.y, v.z, v.w};

  __shared__ float s1[8], s2[8];
  float s  = arr[0] + arr[1] + arr[2] + arr[3];
  float ss = arr[0]*arr[0] + arr[1]*arr[1] + arr[2]*arr[2] + arr[3]*arr[3];
  float mu  = block_sum(s,  s1) * (1.0f / C_);
  float var = block_sum(ss, s2) * (1.0f / C_) - mu * mu;
  float rs  = rsqrtf(var + EPS_);

  const int i0 = threadIdx.x * 4;
  #pragma unroll
  for (int j = 0; j < 4; ++j) {
    int c = i0 + j;
    float y = (arr[j] - mu) * rs * w[c] + bb[c];
    xk[base + c] = (_Float16)(y + pos[base + c]);
    int h = c >> 6, d = c & 63;
    vt[(((size_t)bidx * H_ + h) * D_ + d) * NK_ + kk] = (_Float16)y;
  }
}

// ---------------------------------------------------------------------------
// Kernel 3: Wk f32 -> f16
// ---------------------------------------------------------------------------
__global__ void cvt_wk_kernel(const float* __restrict__ in,
                              _Float16* __restrict__ out) {
  const int i = blockIdx.x * blockDim.x + threadIdx.x;  // 262144 threads
  float4 v = ((const float4*)in)[i];
  out[4*i + 0] = (_Float16)v.x;
  out[4*i + 1] = (_Float16)v.y;
  out[4*i + 2] = (_Float16)v.z;
  out[4*i + 3] = (_Float16)v.w;
}

// ---------------------------------------------------------------------------
// Kernel 4: K projection GEMM  kp[m][j] = sum_c xk[m][c] * Wk[j][c]
// M=32768, N=1024, K=1024.
// 256 threads / 8 waves per block; block tile 128x64, BK=64.
// Double-buffered LDS staged with GLOBAL_LOAD_ASYNC_TO_LDS_B128 (ASYNCcnt).
// Wave w owns m-strip w (16 rows) x all 64 columns (4 accumulators).
// ---------------------------------------------------------------------------
constexpr int BM_ = 128, BN_ = 64, BK_ = 64;

__global__ void kproj_kernel(const _Float16* __restrict__ A,
                             const _Float16* __restrict__ Bw,
                             _Float16* __restrict__ Out) {
  __shared__ _Float16 lA[2][BM_ * BK_];   // 16 KB per buffer
  __shared__ _Float16 lB[2][BN_ * BK_];   //  8 KB per buffer

  const int tid  = threadIdx.x;
  const int lane = tid & 31;
  const int wid  = tid >> 5;
  const int mblk = blockIdx.x >> 4;          // 256 m-blocks
  const int nblk = blockIdx.x & 15;          // 16 n-blocks
  const int m0 = mblk * BM_, n0 = nblk * BN_;
  const int l15 = lane & 15;
  const int koA = (lane >> 4) * 8;
  const int koB = (lane >> 4) * 8;

  // Issue one stage of async copies: 6 x 16B per thread (4 for A, 2 for B).
  auto stage = [&](int buf, int k0) {
    #pragma unroll
    for (int i = 0; i < 4; ++i) {
      int ch = tid + i * 256;                // 1024 chunks for A (128 x 64h)
      int r = ch >> 3, c = (ch & 7) * 8;
      async_copy16(A + (size_t)(m0 + r) * C_ + k0 + c, &lA[buf][r * BK_ + c]);
    }
    #pragma unroll
    for (int i = 0; i < 2; ++i) {
      int ch = tid + i * 256;                // 512 chunks for B (64 x 64h)
      int r = ch >> 3, c = (ch & 7) * 8;
      async_copy16(Bw + (size_t)(n0 + r) * C_ + k0 + c, &lB[buf][r * BK_ + c]);
    }
  };

  v8f acc[4] = {};
  stage(0, 0);
  constexpr int NSTAGE = C_ / BK_;           // 16
  for (int s = 0; s < NSTAGE; ++s) {
    const int cur = s & 1;
    if (s + 1 < NSTAGE) {
      stage(cur ^ 1, (s + 1) * BK_);         // prefetch next stage
      WAIT_ASYNC(6);                         // wait for stage s only
    } else {
      WAIT_ASYNC(0);
    }
    __syncthreads();

    const _Float16* Ar = &lA[cur][(wid * 16 + l15) * BK_];
    #pragma unroll
    for (int kk = 0; kk < BK_; kk += 32) {
      v16h a = load_fragA(Ar + kk + koA);
      #pragma unroll
      for (int t = 0; t < 4; ++t) {
        const _Float16* Br = &lB[cur][(t * 16 + l15) * BK_ + kk + koB];
        acc[t] = wmma_f16(a, load_fragB(Br), acc[t]);
      }
    }
    __syncthreads();                         // protect buffer before reuse
  }

  const int mb = (lane >> 4) * 8;
  #pragma unroll
  for (int t = 0; t < 4; ++t)
    #pragma unroll
    for (int r = 0; r < 8; ++r)
      Out[(size_t)(m0 + wid * 16 + mb + r) * C_ + n0 + t * 16 + l15] =
          (_Float16)acc[t][r];
}

// ---------------------------------------------------------------------------
// Kernel 5: scores = scale*Q@K^T + bias, mask, softmax -> attn (f32, one write)
// One block per (b,h, 16 query rows). 8 waves. 16x4096 f32 stripe in 256KB LDS.
// B fragments register-double-buffered to hide global latency.
// ---------------------------------------------------------------------------
__global__ void scores_softmax_kernel(const _Float16* __restrict__ Q,
                                      const _Float16* __restrict__ K,
                                      const float* __restrict__ bias,
                                      const unsigned char* __restrict__ kpm,
                                      float* __restrict__ attn) {
  extern __shared__ float sc[];              // [16][4096]
  const int q0   = blockIdx.x * 16;
  const int bh   = blockIdx.y;               // b*H + h
  const int bidx = bh >> 4;
  const int h    = bh & 15;
  const int lane = threadIdx.x & 31;
  const int wid  = threadIdx.x >> 5;
  const int l15  = lane & 15;
  const int koA  = (lane >> 4) * 8;
  const int koB  = (lane >> 4) * 8;

  // Q fragments for this 16-row tile (D=64 -> two k-steps), reused for all N.
  const _Float16* qrow = Q + (size_t)(bidx * NQ_ + q0 + l15) * C_ + h * D_;
  v16h a0 = load_fragA(qrow + 0  + koA);
  v16h a1 = load_fragA(qrow + 32 + koA);

  const _Float16* kbase =
      K + (size_t)(bidx * NK_ + l15) * C_ + h * D_;   // + n0*C_ per tile
  const int mb = (lane >> 4) * 8;
  constexpr int NT = NK_ / 16;               // 256 n-tiles, 32 per wave

  v16h b0 = load_fragB(kbase + (size_t)(wid * 16) * C_ + koB);
  v16h b1 = load_fragB(kbase + (size_t)(wid * 16) * C_ + 32 + koB);
  for (int nt = wid; nt < NT; nt += 8) {
    const int pn = (nt + 8 < NT) ? nt + 8 : nt;      // prefetch next tile
    v16h nb0 = load_fragB(kbase + (size_t)(pn * 16) * C_ + koB);
    v16h nb1 = load_fragB(kbase + (size_t)(pn * 16) * C_ + 32 + koB);

    v8f acc = {};
    acc = wmma_f16(a0, b0, acc);
    acc = wmma_f16(a1, b1, acc);

    const int n = nt * 16 + l15;
    const bool masked = kpm[(size_t)bidx * NK_ + n] != 0;
    #pragma unroll
    for (int r = 0; r < 8; ++r) {
      const int m = mb + r;
      float v = acc[r] * SCALE_ +
                bias[((size_t)bh * NQ_ + q0 + m) * NK_ + n];
      sc[m * NK_ + n] = masked ? -1e9f : v;
    }
    b0 = nb0; b1 = nb1;
  }
  __syncthreads();

  // Softmax: each wave owns 2 rows; write attn exactly once.
  for (int rr = wid; rr < 16; rr += 8) {
    float* rp = sc + rr * NK_;
    float mx = -1e30f;
    for (int i = lane; i < NK_; i += 32) mx = fmaxf(mx, rp[i]);
    mx = wave_max(mx);
    float s = 0.f;
    for (int i = lane; i < NK_; i += 32) {
      float e = __expf(rp[i] - mx);
      rp[i] = e;
      s += e;
    }
    s = wave_sum(s);
    const float inv = 1.0f / s;
    float* dst = attn + ((size_t)bh * NQ_ + q0 + rr) * NK_;
    for (int i = lane; i < NK_; i += 32) dst[i] = rp[i] * inv;
  }
}

// ---------------------------------------------------------------------------
// Kernel 6: out = attn @ V (per head) + residual. One block per (b,h,16 q rows),
// 4 waves, each wave owns a 16x16 d-tile; K loop over NK=4096 with register
// double buffering of both operands.
// ---------------------------------------------------------------------------
__global__ void av_residual_kernel(const float* __restrict__ attn,
                                   const _Float16* __restrict__ Vt,
                                   const float* __restrict__ query,
                                   float* __restrict__ out) {
  const int q0   = blockIdx.x * 16;
  const int bh   = blockIdx.y;
  const int bidx = bh >> 4;
  const int h    = bh & 15;
  const int lane = threadIdx.x & 31;
  const int wid  = threadIdx.x >> 5;     // 4 waves -> d-tiles
  const int l15  = lane & 15;
  const int koA  = (lane >> 4) * 8;
  const int koB  = (lane >> 4) * 8;
  const int d0   = wid * 16;

  const float*    arow = attn + ((size_t)bh * NQ_ + q0 + l15) * NK_;
  const _Float16* vrow = Vt + ((size_t)bh * D_ + d0 + l15) * NK_;

  v8f acc = {};
  v8f lo = *(const v8f*)(arow + koA);
  v8f hi = *(const v8f*)(arow + koA + 16);
  v16h b = load_fragB(vrow + koB);
  for (int kb = 0; kb < NK_; kb += 32) {
    const int nk = (kb + 32 < NK_) ? kb + 32 : kb;   // prefetch next k-step
    v8f  nlo = *(const v8f*)(arow + nk + koA);
    v8f  nhi = *(const v8f*)(arow + nk + koA + 16);
    v16h nb  = load_fragB(vrow + nk + koB);

    v16h a;
    #pragma unroll
    for (int i = 0; i < 8; ++i) {
      a[i]     = (_Float16)lo[i];
      a[i + 8] = (_Float16)hi[i];
    }
    acc = wmma_f16(a, b, acc);
    lo = nlo; hi = nhi; b = nb;
  }

  const int mb = (lane >> 4) * 8;
  #pragma unroll
  for (int r = 0; r < 8; ++r) {
    size_t idx = ((size_t)bidx * NQ_ + q0 + mb + r) * C_ + h * D_ + d0 + l15;
    out[idx] = query[idx] + acc[r];
  }
}

// ---------------------------------------------------------------------------
// Launch
// ---------------------------------------------------------------------------
extern "C" void kernel_launch(void* const* d_in, const int* in_sizes, int n_in,
                              void* d_out, int out_size, void* d_ws, size_t ws_size,
                              hipStream_t stream) {
  const float* query  = (const float*)d_in[0];
  const float* memory = (const float*)d_in[1];
  const float* qpos   = (const float*)d_in[2];
  const float* mpos   = (const float*)d_in[3];
  const float* bias   = (const float*)d_in[4];
  const unsigned char* kpm = (const unsigned char*)d_in[5];  // bool mask
  const float* Wk     = (const float*)d_in[6];
  const float* lnqw   = (const float*)d_in[7];
  const float* lnqb   = (const float*)d_in[8];
  const float* lnkw   = (const float*)d_in[9];
  const float* lnkb   = (const float*)d_in[10];

  float* out  = (float*)d_out;                       // [B,NQ,C]
  float* attn = out + (size_t)B_ * NQ_ * C_;         // [B,H,NQ,NK]

  // Workspace layout (f16 intermediates)
  char* ws = (char*)d_ws;
  _Float16* q16  = (_Float16*)ws; ws += (size_t)B_ * NQ_ * C_ * 2;   //  4 MB
  _Float16* xk16 = (_Float16*)ws; ws += (size_t)B_ * NK_ * C_ * 2;   // 64 MB
  _Float16* wk16 = (_Float16*)ws; ws += (size_t)C_ * C_ * 2;         //  2 MB
  _Float16* kp16 = (_Float16*)ws; ws += (size_t)B_ * NK_ * C_ * 2;   // 64 MB
  _Float16* vt16 = (_Float16*)ws;                                     // 64 MB

  prep_q_kernel<<<B_ * NQ_, 256, 0, stream>>>(query, qpos, lnqw, lnqb, q16);
  prep_k_kernel<<<B_ * NK_, 256, 0, stream>>>(memory, mpos, lnkw, lnkb, xk16, vt16);
  cvt_wk_kernel<<<(C_ * C_ / 4) / 256, 256, 0, stream>>>(Wk, wk16);

  // 256 m-blocks * 16 n-blocks
  kproj_kernel<<<4096, 256, 0, stream>>>(xk16, wk16, kp16);

  dim3 g5(NQ_ / 16, B_ * H_);
  scores_softmax_kernel<<<g5, 256, 16 * NK_ * sizeof(float), stream>>>(
      q16, kp16, bias, kpm, attn);

  dim3 g6(NQ_ / 16, B_ * H_);
  av_residual_kernel<<<g6, 128, 0, stream>>>(attn, vt16, query, out);
}
